// STFT_12584254177799
// MI455X (gfx1250) — matrix operand order repaired and checked by reading, not compiled, not run on hardware
//
#include <hip/hip_runtime.h>

// ---------------- problem constants ----------------
#define L_IN      1280000          // samples per batch
#define NBATCH    16
#define WINLEN    1024
#define HOP       512
#define PADR      512
#define LPAD      (L_IN + 2*PADR)  // 1281024 reflect-padded length
#define NFRAMES   2501
#define DIMR      513              // rfft bins
#define MROWS     1026             // real(513) + imag(513)

// ---------------- tiling ----------------
#define MT        2                // M-tiles (of 16 rows) per wave
#define NT        4                // N-tiles (of 16 frames) per block (shared via LDS)
#define WAVES_PER_BLOCK 4
#define MT_PER_BLOCK (WAVES_PER_BLOCK*MT)          // 8 M-tiles per block
#define MSUPER    9                // ceil(65 / 8) m-super-groups
#define MTILES_PAD (MSUPER*MT_PER_BLOCK)           // 72 (tiles >=65 are zero pad)
#define NGROUPS   40               // ceil(ceil(2501/16)=157 / NT)
#define NFRAMES_PAD (NGROUPS*NT*16)                // 2560
#define XPL       ((NFRAMES_PAD-1)*HOP + WINLEN)   // 1311232 (zero tail past LPAD)
#define KC        32               // 1024 / 32 K-chunks
#define LDS_SLOT  48               // 32B payload + 16B pad: 16B-aligned, fewer bank conflicts

typedef __attribute__((ext_vector_type(16))) __bf16 v16bf;
typedef __attribute__((ext_vector_type(8)))  float  v8f;

union V16U { uint4 q[2]; v16bf v; };

__device__ __forceinline__ unsigned short f32_to_bf16(float f) {
    unsigned int u = __float_as_uint(f);
    u += 0x7fffu + ((u >> 16) & 1u);          // round-to-nearest-even
    return (unsigned short)(u >> 16);
}
__device__ __forceinline__ float bf16_to_f32(unsigned short h) {
    return __uint_as_float(((unsigned int)h) << 16);
}

// ---- pre-pass 1: reflect-pad x, split into bf16 hi/lo planes ----
__global__ void stft_prep_x(const float* __restrict__ x,
                            unsigned short* __restrict__ xp_hi,
                            unsigned short* __restrict__ xp_lo)
{
    unsigned int idx = blockIdx.x * blockDim.x + threadIdx.x;
    if (idx >= (unsigned)NBATCH * (unsigned)XPL) return;
    const unsigned int b = idx / (unsigned)XPL;
    const int i = (int)(idx - b * (unsigned)XPL);
    float v = 0.0f;
    if (i < LPAD) {
        int j = i - PADR;
        if (j < 0) j = -j;                         // left reflect (no edge repeat)
        else if (j >= L_IN) j = 2 * L_IN - 2 - j;  // right reflect
        v = x[(size_t)b * L_IN + j];
    }
    const unsigned short hi = f32_to_bf16(v);
    const float lo = v - bf16_to_f32(hi);
    xp_hi[idx] = hi;
    xp_lo[idx] = f32_to_bf16(lo);
}

// ---- pre-pass 2: fft_k -> WMMA A-operand swizzle, bf16 hi/lo ----
// layout: ((mt*KC + kc)*32 + lane)*16 + slot; 16-bit A 16x32 VGPR map:
// slot<8 -> K = kc*32 + g*8 + slot ; slot>=8 -> K = kc*32 + 16 + g*8 + (slot-8)
// (g = lane>>4, M row = mt*16 + (lane&15)); rows >= 1026 are zero.
#define AK_ELEMS  ((size_t)MTILES_PAD * KC * 32 * 16)   // 72*16384 = 1,179,648
__global__ void stft_prep_ak(const float* __restrict__ fk,
                             unsigned short* __restrict__ ak_hi,
                             unsigned short* __restrict__ ak_lo)
{
    unsigned int idx = blockIdx.x * blockDim.x + threadIdx.x;
    if (idx >= (unsigned)AK_ELEMS) return;
    const int slot = idx & 15;
    const int lane = (idx >> 4) & 31;
    const int kc   = (idx >> 9) & 31;
    const int mt   = idx >> 14;
    const int g = lane >> 4;
    const int m = mt * 16 + (lane & 15);
    const int k = kc * 32 + ((slot < 8) ? (g * 8 + slot) : (16 + g * 8 + (slot - 8)));
    const float v = (m < MROWS) ? fk[(size_t)m * WINLEN + k] : 0.0f;
    const unsigned short hi = f32_to_bf16(v);
    const float lo = v - bf16_to_f32(hi);
    ak_hi[idx] = hi;
    ak_lo[idx] = f32_to_bf16(lo);
}

// ---- main kernel: 4 waves/block; B tiles staged in LDS, shared by all waves ----
__global__ __launch_bounds__(128)
void stft_wmma_gemm(const unsigned short* __restrict__ xp_hi,
                    const unsigned short* __restrict__ xp_lo,
                    const unsigned short* __restrict__ ak_hi,
                    const unsigned short* __restrict__ ak_lo,
                    float* __restrict__ out)
{
    // [buf][plane*NT + ntile][lane][48B slot, 32B payload]   -> 24 KB
    __shared__ __align__(16) unsigned char smem[2][2 * NT][32][LDS_SLOT];

    const int tid  = (int)threadIdx.x;
    const int lane = tid & 31;
    const int wv   = tid >> 5;             // wave id 0..3
    const int lmod = lane & 15;
    const int lg   = lane >> 4;
    const int ng   = (int)blockIdx.x;
    const int mg   = (int)blockIdx.y;
    const int b    = (int)blockIdx.z;
    const int n0   = ng * (NT * 16);
    const int mt0  = mg * MT_PER_BLOCK + wv * MT;

    const unsigned short* xb_hi = xp_hi + (size_t)b * XPL;
    const unsigned short* xb_lo = xp_lo + (size_t)b * XPL;

    // Each thread stages 4x16B chunks of B per K-step (128 thr * 64B = 8KB = hi+lo).
    const unsigned short* gsrc[4];
    unsigned int          ldst[4];
#pragma unroll
    for (int q = 0; q < 4; ++q) {
        const int c  = tid * 4 + q;        // 0..511 chunk id
        const int p  = c >> 8;             // plane: 0=hi, 1=lo
        const int t  = (c >> 6) & (NT - 1);
        const int ln = (c >> 1) & 31;      // destination lane
        const int h  = c & 1;              // which 16B half of the 32B operand
        const unsigned short* base = p ? xb_lo : xb_hi;
        gsrc[q] = base + (size_t)(n0 + t * 16 + (ln & 15)) * HOP
                       + (size_t)(ln >> 4) * 8 + (size_t)h * 16;
        ldst[q] = (unsigned int)(((p * NT + t) * 32 + ln) * LDS_SLOT + h * 16);
    }

    v8f acc[MT][NT];
#pragma unroll
    for (int i = 0; i < MT; ++i)
#pragma unroll
        for (int t = 0; t < NT; ++t)
#pragma unroll
            for (int r = 0; r < 8; ++r) acc[i][t][r] = 0.0f;

    size_t a_base[MT];
#pragma unroll
    for (int i = 0; i < MT; ++i)
        a_base[i] = (((size_t)(mt0 + i) * KC) * 32 + (size_t)lane) * 16;

    unsigned char* const sbase = &smem[0][0][0][0];
    const unsigned int bufstride = 2 * NT * 32 * LDS_SLOT;   // 12288 bytes

    // prologue: stage K-chunk 0 into buffer 0
    uint4 st[4];
#pragma unroll
    for (int q = 0; q < 4; ++q) st[q] = *reinterpret_cast<const uint4*>(gsrc[q]);
#pragma unroll
    for (int q = 0; q < 4; ++q) *reinterpret_cast<uint4*>(sbase + ldst[q]) = st[q];
    __syncthreads();

    for (int kc = 0; kc < KC; ++kc) {
        const int cur = kc & 1;
        // issue next K-chunk's global loads early (a full compute section of latency)
        if (kc + 1 < KC) {
#pragma unroll
            for (int q = 0; q < 4; ++q)
                st[q] = *reinterpret_cast<const uint4*>(gsrc[q] + (size_t)(kc + 1) * 32);
        }

        v16bf ahi[MT], alo[MT];
#pragma unroll
        for (int i = 0; i < MT; ++i) {
            const size_t ao = a_base[i] + (size_t)kc * (32 * 16);
            ahi[i] = *reinterpret_cast<const v16bf*>(ak_hi + ao);
            alo[i] = *reinterpret_cast<const v16bf*>(ak_lo + ao);
        }

        unsigned char* sb = sbase + (unsigned int)cur * bufstride;
#pragma unroll
        for (int t = 0; t < NT; ++t) {
            V16U bh, bl;
            const unsigned int ohi = (unsigned int)(((0 * NT + t) * 32 + lane) * LDS_SLOT);
            const unsigned int olo = (unsigned int)(((1 * NT + t) * 32 + lane) * LDS_SLOT);
            bh.q[0] = *reinterpret_cast<const uint4*>(sb + ohi);
            bh.q[1] = *reinterpret_cast<const uint4*>(sb + ohi + 16);
            bl.q[0] = *reinterpret_cast<const uint4*>(sb + olo);
            bl.q[1] = *reinterpret_cast<const uint4*>(sb + olo + 16);
#pragma unroll
            for (int i = 0; i < MT; ++i) {
                // split-precision product: hi*hi + lo*hi + hi*lo  (~fp32 accuracy)
                acc[i][t] = __builtin_amdgcn_wmma_f32_16x16x32_bf16(
                    false, ahi[i], false, bh.v, (short)0, acc[i][t], false, false);
                acc[i][t] = __builtin_amdgcn_wmma_f32_16x16x32_bf16(
                    false, alo[i], false, bh.v, (short)0, acc[i][t], false, false);
                acc[i][t] = __builtin_amdgcn_wmma_f32_16x16x32_bf16(
                    false, ahi[i], false, bl.v, (short)0, acc[i][t], false, false);
            }
        }

        // write next K-chunk into the other buffer (its readers synced last iter)
        if (kc + 1 < KC) {
            unsigned char* nb = sbase + (unsigned int)(1 - cur) * bufstride;
#pragma unroll
            for (int q = 0; q < 4; ++q) *reinterpret_cast<uint4*>(nb + ldst[q]) = st[q];
        }
        __syncthreads();
    }

    // store: C/D layout — VGPR r holds (M = r + 8*lg, N = lane&15)
    const size_t imag_off = (size_t)NBATCH * DIMR * NFRAMES;   // 20,528,208
#pragma unroll
    for (int i = 0; i < MT; ++i) {
        const int mb = (mt0 + i) * 16 + lg * 8;
#pragma unroll
        for (int r = 0; r < 8; ++r) {
            const int m = mb + r;
            if (m < MROWS) {
                float* orow = (m < DIMR)
                    ? out + ((size_t)b * DIMR + m) * NFRAMES
                    : out + imag_off + ((size_t)b * DIMR + (m - DIMR)) * NFRAMES;
#pragma unroll
                for (int t = 0; t < NT; ++t) {
                    const int f = n0 + t * 16 + lmod;
                    if (f < NFRAMES) orow[f] = acc[i][t][r];
                }
            }
        }
    }
}

extern "C" void kernel_launch(void* const* d_in, const int* in_sizes, int n_in,
                              void* d_out, int out_size, void* d_ws, size_t ws_size,
                              hipStream_t stream)
{
    const float* x  = (const float*)d_in[0];   // (16, 1280000) f32
    const float* fk = (const float*)d_in[1];   // (1026, 1024) f32
    float* out = (float*)d_out;                // real(16,513,2501) ++ imag(16,513,2501)

    const size_t XP_BYTES = (size_t)NBATCH * XPL * sizeof(unsigned short); // 41,959,424
    const size_t AK_BYTES = AK_ELEMS * sizeof(unsigned short);             //  2,359,296
    const size_t WS_NEED  = 2 * XP_BYTES + 2 * AK_BYTES;                   // ~88.6 MB
    if (ws_size < WS_NEED) return;   // defensive: avoid corrupting memory

    char* ws = (char*)d_ws;
    unsigned short* xp_hi = (unsigned short*)(ws);
    unsigned short* xp_lo = (unsigned short*)(ws + XP_BYTES);
    unsigned short* ak_hi = (unsigned short*)(ws + 2 * XP_BYTES);
    unsigned short* ak_lo = (unsigned short*)(ws + 2 * XP_BYTES + AK_BYTES);

    const unsigned xtotal = (unsigned)NBATCH * (unsigned)XPL;              // 20,979,712
    stft_prep_x<<<(xtotal + 255u) / 256u, 256, 0, stream>>>(x, xp_hi, xp_lo);
    stft_prep_ak<<<((unsigned)AK_ELEMS + 255u) / 256u, 256, 0, stream>>>(fk, ak_hi, ak_lo);

    dim3 grid(NGROUPS, MSUPER, NBATCH);   // 40 x 9 x 16 blocks of 4 waves
    stft_wmma_gemm<<<grid, 128, 0, stream>>>(xp_hi, xp_lo, ak_hi, ak_lo, out);
}
